// AttentionHead_82875688943918
// MI455X (gfx1250) — compile-verified, matrix operand order checked by reading, not run on hardware
//
#include <hip/hip_runtime.h>
#include <hip/hip_bf16.h>

// ---------------------------------------------------------------------------
// Types for CDNA5 WMMA (wave32): v_wmma_f32_16x16x32_bf16
// ---------------------------------------------------------------------------
typedef __attribute__((ext_vector_type(16))) __bf16 v16bf;
typedef __attribute__((ext_vector_type(8)))  __bf16 v8bf;
typedef __attribute__((ext_vector_type(8)))  float  v8f;

union ABu {
    v16bf v;
    v8bf  h[2];
};

__device__ __forceinline__ __bf16 f2bf(float f) {
    return (__bf16)f;               // native f32->bf16 conversion path
}

__device__ __forceinline__ v8bf pack8(float4 a, float4 b) {
    v8bf r;
    r[0] = f2bf(a.x); r[1] = f2bf(a.y); r[2] = f2bf(a.z); r[3] = f2bf(a.w);
    r[4] = f2bf(b.x); r[5] = f2bf(b.y); r[6] = f2bf(b.z); r[7] = f2bf(b.w);
    return r;
}

__device__ __forceinline__ v8f wmma_bf16(const ABu& a, const ABu& b, v8f c) {
    return __builtin_amdgcn_wmma_f32_16x16x32_bf16(
        /*neg_a=*/false, a.v, /*neg_b=*/false, b.v,
        /*c_mod=*/(short)0, c, /*reuse_a=*/false, /*reuse_b=*/false);
}

#define EMBED 1024
#define HEADD 64
#define SEQT  4096
#define NB    4

// ---------------------------------------------------------------------------
// Kernel 1: QKV projection.  grid = (M/256, 3), block = 256 (8 waves).
// Each wave computes a 32x64 output tile (2 A-frags sharing 4 B-frags ->
// 8 WMMAs per K-step, each B-frag reused twice).  Weights staged into LDS
// transposed+padded bf16 so B-frags are contiguous ds_load_b128.
// Q,K stored row-major bf16; V stored transposed [b][d][t] bf16.
// ---------------------------------------------------------------------------
#define EBLK 256
#define WPAD 8
#define WSTRIDE (EBLK + WPAD)   // bf16 elements per d-row in LDS

__global__ void __launch_bounds__(256)
qkv_proj_kernel(const float* __restrict__ x,
                const float* __restrict__ Wq,
                const float* __restrict__ Wk,
                const float* __restrict__ Wv,
                __bf16* __restrict__ Q,
                __bf16* __restrict__ K,
                __bf16* __restrict__ Vt)
{
    __shared__ __attribute__((aligned(16))) __bf16 ldsW[HEADD * WSTRIDE];

    const int z = blockIdx.y;                       // 0=Q, 1=K, 2=V
    const float* __restrict__ W = (z == 0) ? Wq : (z == 1) ? Wk : Wv;

    const int t    = threadIdx.x;
    const int wave = t >> 5;
    const int lane = t & 31;
    const int lq   = lane & 15;
    const int ksel = (lane >> 4) << 3;              // 0 or 8 (K sub-select)
    const int mBase = blockIdx.x * 256 + wave * 32;
    const int row0  = mBase + lq;                   // A-frag 0 row
    const int row1  = mBase + 16 + lq;              // A-frag 1 row

    v8f acc[2][4] = {{v8f{}, v8f{}, v8f{}, v8f{}},
                     {v8f{}, v8f{}, v8f{}, v8f{}}};

    for (int blk = 0; blk < EMBED / EBLK; ++blk) {
        const int ebase = blk * EBLK;

        // ---- stage EBLK x 64 weight chunk into LDS, transposed, bf16 ----
        #pragma unroll 4
        for (int it = 0; it < (EBLK * HEADD) / (256 * 4); ++it) {
            const int idx  = it * 1024 + t * 4;     // flat idx within chunk
            const int eloc = idx >> 6;
            const int d    = idx & 63;
            float4 w4 = *(const float4*)(W + ebase * HEADD + idx);
            ldsW[(d + 0) * WSTRIDE + eloc] = f2bf(w4.x);
            ldsW[(d + 1) * WSTRIDE + eloc] = f2bf(w4.y);
            ldsW[(d + 2) * WSTRIDE + eloc] = f2bf(w4.z);
            ldsW[(d + 3) * WSTRIDE + eloc] = f2bf(w4.w);
        }
        __syncthreads();

        // ---- 8 WMMA K-steps over this chunk ----
        for (int kk = 0; kk < EBLK; kk += 32) {
            const int kg = ebase + kk;

            ABu a0, a1;
            {
                const float* xr0 = x + row0 * EMBED + kg + ksel;
                a0.h[0] = pack8(*(const float4*)(xr0 + 0),  *(const float4*)(xr0 + 4));
                a0.h[1] = pack8(*(const float4*)(xr0 + 16), *(const float4*)(xr0 + 20));
                const float* xr1 = x + row1 * EMBED + kg + ksel;
                a1.h[0] = pack8(*(const float4*)(xr1 + 0),  *(const float4*)(xr1 + 4));
                a1.h[1] = pack8(*(const float4*)(xr1 + 16), *(const float4*)(xr1 + 20));
            }

            // load all 4 B-fragments first so dscnt waits amortize
            ABu bfr[4];
            #pragma unroll
            for (int n = 0; n < 4; ++n) {
                const int d = n * 16 + lq;
                bfr[n].h[0] = *(const v8bf*)(ldsW + d * WSTRIDE + kk + ksel);
                bfr[n].h[1] = *(const v8bf*)(ldsW + d * WSTRIDE + kk + ksel + 16);
            }
            #pragma unroll
            for (int n = 0; n < 4; ++n)
                acc[0][n] = wmma_bf16(a0, bfr[n], acc[0][n]);
            #pragma unroll
            for (int n = 0; n < 4; ++n)
                acc[1][n] = wmma_bf16(a1, bfr[n], acc[1][n]);
        }
        __syncthreads();
    }

    // ---- epilogue: C layout -> bf16 stores ----
    const int rsel = (lane >> 4) << 3;              // C row offset 0 / 8
    #pragma unroll
    for (int r = 0; r < 2; ++r) {
        #pragma unroll
        for (int n = 0; n < 4; ++n) {
            const int d = n * 16 + lq;
            #pragma unroll
            for (int j = 0; j < 8; ++j) {
                const int m = mBase + r * 16 + j + rsel;
                const __bf16 val = f2bf(acc[r][n][j]);
                if (z == 0) {
                    Q[m * HEADD + d] = val;
                } else if (z == 1) {
                    K[m * HEADD + d] = val;
                } else {
                    const int b  = m >> 12;         // m / 4096
                    const int tt = m & 4095;
                    Vt[((b * HEADD + d) << 12) + tt] = val;
                }
            }
        }
    }
}

// ---------------------------------------------------------------------------
// Kernel 2: flash attention.  grid = (T/128, B), block = 256 (8 waves).
// Per 32-key block: K/V tiles (4 KB each) are staged cooperatively into LDS
// (one global_load_b128 + ds_store_b128 per thread), next block is
// prefetched (global_prefetch_b8), then each wave runs
//   S = Q K^T (4 WMMAs) -> scale/mask -> online softmax (shfl_xor reduce)
//   P -> per-wave LDS transpose -> O += P V (4 WMMAs)
// ---------------------------------------------------------------------------
__global__ void __launch_bounds__(256)
flash_attn_kernel(const __bf16* __restrict__ Qg,
                  const __bf16* __restrict__ Kg,
                  const __bf16* __restrict__ Vtg,
                  float* __restrict__ out)
{
    __shared__ __attribute__((aligned(16))) __bf16 ldsK[32 * HEADD];   // [key][d]
    __shared__ __attribute__((aligned(16))) __bf16 ldsV[HEADD * 32];   // [d][key]
    __shared__ __attribute__((aligned(16))) __bf16 ldsP[8 * 16 * 32];  // per-wave P

    const int b    = blockIdx.y;
    const int tile = blockIdx.x;
    const int t    = threadIdx.x;
    const int wave = t >> 5;
    const int lane = t & 31;
    const int lq   = lane & 15;
    const int sel  = (lane >> 4) << 3;              // 0 or 8
    const int qBase = tile * 128 + wave * 16;

    const __bf16* __restrict__ Qb = Qg  + b * SEQT * HEADD;
    const __bf16* __restrict__ Kb = Kg  + b * SEQT * HEADD;
    const __bf16* __restrict__ Vb = Vtg + b * HEADD * SEQT;

    // staging coordinates (per thread, 16B chunks)
    const int skK = t >> 3;                         // key row 0..31
    const int sdK = (t & 7) * 8;                    // d offset 0..56
    const int sdV = t >> 2;                         // d row 0..63
    const int skV = (t & 3) * 8;                    // key offset 0..24

    // Q A-fragments (16 x 64 = two 16x32 frags), loaded once
    ABu qa[2];
    {
        const __bf16* qr = Qb + (qBase + lq) * HEADD;
        qa[0].h[0] = *(const v8bf*)(qr + sel);
        qa[0].h[1] = *(const v8bf*)(qr + 16 + sel);
        qa[1].h[0] = *(const v8bf*)(qr + 32 + sel);
        qa[1].h[1] = *(const v8bf*)(qr + 48 + sel);
    }

    v8f o[4] = {v8f{}, v8f{}, v8f{}, v8f{}};
    float m[8], l[8];
    #pragma unroll
    for (int j = 0; j < 8; ++j) { m[j] = -INFINITY; l[j] = 0.0f; }

    const float scale = 0.03125f;                   // 1/sqrt(1024)
    const int kbMax = tile * 4 + 4;                 // key blocks of 32 (uniform per block)

    for (int kb = 0; kb < kbMax; ++kb) {
        const int k0 = kb * 32;

        // ---- cooperative staging of K/V block into LDS ----
        __syncthreads();                            // previous block fully consumed
        {
            v8bf kv = *(const v8bf*)(Kb + (k0 + skK) * HEADD + sdK);
            *(v8bf*)(ldsK + skK * HEADD + sdK) = kv;
            v8bf vv = *(const v8bf*)(Vb + sdV * SEQT + k0 + skV);
            *(v8bf*)(ldsV + sdV * 32 + skV) = vv;
        }
        if (kb + 1 < kbMax) {                       // prefetch next block
            __builtin_prefetch((const void*)(Kb + (k0 + 32 + skK) * HEADD + sdK), 0, 1);
            __builtin_prefetch((const void*)(Vb + sdV * SEQT + k0 + 32 + skV), 0, 1);
        }
        __syncthreads();

        // ---- S = Q K^T for two 16-key chunks ----
        v8f s[2];
        #pragma unroll
        for (int c = 0; c < 2; ++c) {
            const __bf16* kr = ldsK + (c * 16 + lq) * HEADD;
            ABu b0, b1;
            b0.h[0] = *(const v8bf*)(kr + sel);
            b0.h[1] = *(const v8bf*)(kr + 16 + sel);
            b1.h[0] = *(const v8bf*)(kr + 32 + sel);
            b1.h[1] = *(const v8bf*)(kr + 48 + sel);
            v8f c0 = {};
            c0 = wmma_bf16(qa[0], b0, c0);
            c0 = wmma_bf16(qa[1], b1, c0);
            s[c] = c0;
        }

        // ---- scale + causal mask ----
        #pragma unroll
        for (int c = 0; c < 2; ++c) {
            const int key = k0 + c * 16 + lq;
            #pragma unroll
            for (int j = 0; j < 8; ++j) {
                const int qrow = qBase + j + sel;
                const float v = s[c][j] * scale;
                s[c][j] = (key <= qrow) ? v : -INFINITY;
            }
        }

        // ---- online softmax: row max/sum across 16 lanes sharing a row ----
        float mnew[8];
        #pragma unroll
        for (int j = 0; j < 8; ++j) {
            float v = fmaxf(s[0][j], s[1][j]);
            #pragma unroll
            for (int off = 1; off < 16; off <<= 1)
                v = fmaxf(v, __shfl_xor(v, off, 32));
            mnew[j] = fmaxf(m[j], v);
        }
        #pragma unroll
        for (int c = 0; c < 2; ++c)
            #pragma unroll
            for (int j = 0; j < 8; ++j)
                s[c][j] = __expf(s[c][j] - mnew[j]);
        #pragma unroll
        for (int j = 0; j < 8; ++j) {
            float sum = s[0][j] + s[1][j];
            #pragma unroll
            for (int off = 1; off < 16; off <<= 1)
                sum += __shfl_xor(sum, off, 32);
            const float alpha = __expf(m[j] - mnew[j]);
            l[j] = l[j] * alpha + sum;
            m[j] = mnew[j];
            #pragma unroll
            for (int f = 0; f < 4; ++f) o[f][j] *= alpha;
        }

        // ---- P: C-layout -> A-fragment layout via per-wave LDS ----
        __bf16* P = &ldsP[wave * 16 * 32];
        #pragma unroll
        for (int c = 0; c < 2; ++c)
            #pragma unroll
            for (int j = 0; j < 8; ++j) {
                const int r = j + sel;
                P[r * 32 + c * 16 + lq] = f2bf(s[c][j]);
            }
        // per-wave LDS region: same-wave DS ops are in-order, no barrier needed
        ABu pa;
        {
            const __bf16* pr = P + lq * 32;
            pa.h[0] = *(const v8bf*)(pr + sel);
            pa.h[1] = *(const v8bf*)(pr + 16 + sel);
        }

        // ---- O += P * V  (V staged [d][key]: contiguous over key dim) ----
        ABu vfr[4];
        #pragma unroll
        for (int n = 0; n < 4; ++n) {
            const __bf16* vr = ldsV + (n * 16 + lq) * 32;
            vfr[n].h[0] = *(const v8bf*)(vr + sel);
            vfr[n].h[1] = *(const v8bf*)(vr + 16 + sel);
        }
        #pragma unroll
        for (int n = 0; n < 4; ++n)
            o[n] = wmma_bf16(pa, vfr[n], o[n]);
    }

    // ---- epilogue: O / l -> f32 output ----
    #pragma unroll
    for (int n = 0; n < 4; ++n) {
        const int d = n * 16 + lq;
        #pragma unroll
        for (int j = 0; j < 8; ++j) {
            const int qrow = qBase + j + sel;
            const float inv = 1.0f / l[j];
            out[((long)(b * SEQT + qrow)) * HEADD + d] = o[n][j] * inv;
        }
    }
}

// ---------------------------------------------------------------------------
// Host launcher
// ---------------------------------------------------------------------------
extern "C" void kernel_launch(void* const* d_in, const int* in_sizes, int n_in,
                              void* d_out, int out_size, void* d_ws, size_t ws_size,
                              hipStream_t stream) {
    (void)in_sizes; (void)n_in; (void)out_size; (void)ws_size;

    const float* x  = (const float*)d_in[0];
    const float* Wq = (const float*)d_in[1];
    const float* Wk = (const float*)d_in[2];
    const float* Wv = (const float*)d_in[3];
    float* out = (float*)d_out;

    const size_t perMat = (size_t)NB * SEQT * HEADD;   // 1,048,576 elements
    __bf16* Q  = (__bf16*)d_ws;
    __bf16* K  = Q + perMat;
    __bf16* Vt = K + perMat;                            // 6 MB total bf16

    dim3 g1((NB * SEQT) / 256, 3);
    qkv_proj_kernel<<<g1, 256, 0, stream>>>(x, Wq, Wk, Wv, Q, K, Vt);

    dim3 g2(SEQT / 128, NB);
    flash_attn_kernel<<<g2, 256, 0, stream>>>(Q, K, Vt, out);
}
